// Network_46145128628933
// MI455X (gfx1250) — compile-verified, matrix operand order checked by reading, not run on hardware
//
#include <hip/hip_runtime.h>

// ============================================================================
// PointNet++ part-seg forward for MI455X (gfx1250, wave32, WMMA).
// All MLPs run as one big f16 GEMM (f32 accumulate) per layer; activations are
// stored f16 with K padded to 32 so the WMMA GEMM is fully branchless.
// Weights are pre-packed into the exact B-fragment VGPR layout once per launch.
// Geometry kernels (FPS / ball query / 3-NN) stay f32.
//
// Inputs (setup_inputs dict order): 0 points(4,3,4096) 1 class_labels(4,16),
//  2..5 stem, 6..21 sa1..sa4 (W1,b1,W2,b2), 22..37 fp1..fp4, 38 fin_W1 39 fin_b1
//  40 bn_gamma 41 bn_beta 42 fin_W2 43 fin_b2.
// Output: logits (4,50,4096) ++ g0 (4,32,4096), fp32 flat.
// ============================================================================

#define PNB       4
#define PNN       4096
#define PNOBJ     16
#define PNCLS     50
#define PNNS      32
#define PN_BNEPS  1e-5f

typedef _Float16 h16;
typedef __attribute__((ext_vector_type(16))) _Float16 v16h;
typedef __attribute__((ext_vector_type(8)))  _Float16 v8h;
typedef __attribute__((ext_vector_type(8)))  float    v8f;

static inline int pn_r32(int x) { return (x + 31) & ~31; }

// ---------------------------------------------------------------------------
// Branchless WMMA GEMM: Y[M,Np](f16) = act(X[M,Kp](f16) @ Wp + biasp).
//  - M  multiple of 64 (guaranteed by caller), Kp/Np multiples of 32.
//  - X row-major, zero-padded columns K..Kp-1.
//  - Wp fragment-packed: tile (kt,tn) holds 32 lanes x 16 halves contiguously,
//    lane half h = K sub-bank, per the ISA 16-bit B striping; zero-padded.
//  - One wave computes a 16x32 strip: two N-tiles share one A fragment.
// ---------------------------------------------------------------------------
__global__ void __launch_bounds__(128)
pn2_gemm(const h16* __restrict__ X, const h16* __restrict__ Wp,
         const float* __restrict__ bp, h16* __restrict__ Y,
         int M, int Np, int Kp, int do_relu)
{
    const int lane = threadIdx.x & 31;
    const int wave = threadIdx.x >> 5;
    const int tn0  = blockIdx.x * 2;            // first of two N tiles
    const int tm   = blockIdx.y * 4 + wave;     // M tile (always valid)
    const int h    = lane >> 4;                 // lane half -> K sub-bank
    const int l16  = lane & 15;
    const int ntiles = Np >> 4;
    const int rowA = tm * 16 + l16;

    const h16* xr = X + (size_t)rowA * Kp + h * 8;   // A: two b128 loads / iter
    v8f acc0 = {}, acc1 = {};

    const int kts = Kp >> 5;
    for (int kt = 0; kt < kts; ++kt) {
        const v8h alo = *(const v8h*)(xr + kt * 32);
        const v8h ahi = *(const v8h*)(xr + kt * 32 + 16);
        v16h a;
#pragma unroll
        for (int i = 0; i < 8; ++i) { a[i] = alo[i]; a[8 + i] = ahi[i]; }

        const h16* wt = Wp + (((size_t)kt * ntiles + tn0) * 32 + lane) * 16;
        const v16h b0 = *(const v16h*)(wt);          // tile tn0   (2x b128)
        const v16h b1 = *(const v16h*)(wt + 512);    // tile tn0+1 (2x b128)
        __builtin_prefetch(wt + (size_t)ntiles * 512, 0, 1);  // next K panel

        acc0 = __builtin_amdgcn_wmma_f32_16x16x32_f16(false, a, false, b0,
                                                      (short)0, acc0, false, false);
        acc1 = __builtin_amdgcn_wmma_f32_16x16x32_f16(false, a, false, b1,
                                                      (short)0, acc1, false, false);
    }

    const int c0 = tn0 * 16 + l16;
    const int c1 = c0 + 16;
    const float bb0 = bp[c0], bb1 = bp[c1];
#pragma unroll
    for (int r = 0; r < 8; ++r) {
        const int row = tm * 16 + h * 8 + r;        // C/D: VGPR r -> row r+8h
        float v0 = acc0[r] + bb0;
        float v1 = acc1[r] + bb1;
        if (do_relu) { v0 = fmaxf(v0, 0.f); v1 = fmaxf(v1, 0.f); }
        Y[(size_t)row * Np + c0] = (h16)v0;
        Y[(size_t)row * Np + c1] = (h16)v1;
    }
}

// ---------------------------------------------------------------------------
// Weight packing: W (K,N) f32 row-major -> fragment-packed f16 (Kp x Np), 0-pad.
// out[((kt*ntiles+tn)*32 + lane)*16 + i] = W[kt*32 + kb, tn*16 + (lane&15)]
//   kb = h*8 + i + (i>=8 ? 8 : 0), h = lane>>4.
// ---------------------------------------------------------------------------
__global__ void pn2_pack_w(const float* __restrict__ W, int K, int N, int Kp, int Np,
                           h16* __restrict__ out)
{
    int e = blockIdx.x * blockDim.x + threadIdx.x;
    int total = (Kp >> 5) * (Np >> 4) * 512;
    if (e >= total) return;
    int i    = e & 15;
    int lane = (e >> 4) & 31;
    int tile = e >> 9;
    int tn   = tile % (Np >> 4);
    int kt   = tile / (Np >> 4);
    int hh   = lane >> 4;
    int n    = tn * 16 + (lane & 15);
    int kb   = hh * 8 + i + (i >= 8 ? 8 : 0);
    int k    = kt * 32 + kb;
    out[e] = (k < K && n < N) ? (h16)W[k * N + n] : (h16)0.f;
}

__global__ void pn2_pack_b(const float* __restrict__ b, int N, int Np, float* __restrict__ out)
{
    int i = blockIdx.x * blockDim.x + threadIdx.x;
    if (i >= Np) return;
    out[i] = (i < N) ? b[i] : 0.f;
}

// ---------------------------------------------------------------------------
// (B,3,N) -> xyz0 (B,N,3) f32  and  xyzp (B*N,32) f16 zero-padded (stem GEMM A).
// ---------------------------------------------------------------------------
__global__ void pn2_points_to_xyz(const float* __restrict__ pts, float* __restrict__ xyz,
                                  h16* __restrict__ xyzp, int B, int N)
{
    int t = blockIdx.x * blockDim.x + threadIdx.x;
    if (t >= B * N) return;
    int b = t / N, n = t % N;
    float v[3];
#pragma unroll
    for (int c = 0; c < 3; ++c) {
        v[c] = pts[(b * 3 + c) * N + n];
        xyz[t * 3 + c] = v[c];
    }
    h16* row = xyzp + (size_t)t * 32;
#pragma unroll
    for (int c = 0; c < 32; ++c) row[c] = (c < 3) ? (h16)v[c] : (h16)0.f;
}

// ---------------------------------------------------------------------------
// Farthest point sampling: one block per batch, min-dist array resident in LDS.
// ---------------------------------------------------------------------------
__global__ void __launch_bounds__(256)
pn2_fps(const float* __restrict__ xyz, int N, int npoint, int* __restrict__ idx)
{
    __shared__ float mind[PNN];
    __shared__ float sval[256];
    __shared__ int   sidx[256];
    __shared__ int   sfar;
    const int b = blockIdx.x;
    const int t = threadIdx.x;
    const float* p = xyz + (size_t)b * N * 3;

    for (int i = t; i < N; i += blockDim.x) mind[i] = 1e10f;
    if (t == 0) sfar = 0;
    __syncthreads();

    for (int it = 0; it < npoint; ++it) {
        const int far = sfar;
        if (t == 0) idx[b * npoint + it] = far;
        const float cx = p[far * 3 + 0], cy = p[far * 3 + 1], cz = p[far * 3 + 2];
        float best = -1.f; int bi = 0x7fffffff;
        for (int i = t; i < N; i += blockDim.x) {
            float dx = p[i * 3 + 0] - cx, dy = p[i * 3 + 1] - cy, dz = p[i * 3 + 2] - cz;
            float d = dx * dx + dy * dy + dz * dz;
            float m = fminf(mind[i], d);
            mind[i] = m;
            if (m > best) { best = m; bi = i; }
        }
        sval[t] = best; sidx[t] = bi;
        __syncthreads();
        for (int s = blockDim.x >> 1; s > 0; s >>= 1) {
            if (t < s) {
                float ov = sval[t + s]; int oi = sidx[t + s];
                if (ov > sval[t] || (ov == sval[t] && oi < sidx[t])) { sval[t] = ov; sidx[t] = oi; }
            }
            __syncthreads();
        }
        if (t == 0) sfar = sidx[0];
        __syncthreads();
    }
}

__global__ void pn2_gather_newxyz(const float* __restrict__ xyz, const int* __restrict__ fidx,
                                  int N, int S, float* __restrict__ nxyz, int B)
{
    int e = blockIdx.x * blockDim.x + threadIdx.x;
    if (e >= B * S * 3) return;
    int c = e % 3, s = (e / 3) % S, b = e / (3 * S);
    nxyz[e] = xyz[((size_t)b * N + fidx[b * S + s]) * 3 + c];
}

// Ball query: first nsample in-radius indices (ascending), pad with first.
__global__ void pn2_ball_query(const float* __restrict__ xyz, const float* __restrict__ nxyz,
                               int N, int S, float r2, int* __restrict__ gidx, int B)
{
    int t = blockIdx.x * blockDim.x + threadIdx.x;
    if (t >= B * S) return;
    int b = t / S;
    const float* p = xyz + (size_t)b * N * 3;
    const float cx = nxyz[t * 3 + 0], cy = nxyz[t * 3 + 1], cz = nxyz[t * 3 + 2];
    int* out = gidx + (size_t)t * PNNS;
    int cnt = 0, first = 0;
    for (int i = 0; i < N && cnt < PNNS; ++i) {
        float dx = p[i * 3 + 0] - cx, dy = p[i * 3 + 1] - cy, dz = p[i * 3 + 2] - cz;
        if (dx * dx + dy * dy + dz * dz <= r2) {
            if (cnt == 0) first = i;
            out[cnt++] = i;
        }
    }
    for (; cnt < PNNS; ++cnt) out[cnt] = first;
}

// Grouped GEMM input: X[(b,s,j), 0..Kp) = [rel_xyz(3) | feats(Cin) | 0 pad]
__global__ void pn2_group_build(const float* __restrict__ xyz, const float* __restrict__ nxyz,
                                const h16* __restrict__ feats, const int* __restrict__ gidx,
                                int N, int S, int Cin, int Kp, h16* __restrict__ X, int B)
{
    long long e = (long long)blockIdx.x * blockDim.x + threadIdx.x;
    long long total = (long long)B * S * PNNS * Kp;
    if (e >= total) return;
    int c = (int)(e % Kp);
    long long row = e / Kp;
    int j = (int)(row % PNNS);
    int s = (int)((row / PNNS) % S);
    int b = (int)(row / ((long long)PNNS * S));
    if (c >= Cin + 3) { X[e] = (h16)0.f; return; }
    int gi = gidx[((size_t)(b * S + s)) * PNNS + j];
    if (c < 3)
        X[e] = (h16)(xyz[((size_t)b * N + gi) * 3 + c] - nxyz[(size_t)(b * S + s) * 3 + c]);
    else
        X[e] = feats[((size_t)b * N + gi) * Cin + (c - 3)];
}

// Max-pool over nsample
__global__ void pn2_maxpool(const h16* __restrict__ H, int G, int C, h16* __restrict__ out)
{
    int t = blockIdx.x * blockDim.x + threadIdx.x;
    if (t >= G * C) return;
    int g = t / C, c = t % C;
    float m = -1e30f;
    for (int j = 0; j < PNNS; ++j) m = fmaxf(m, (float)H[((size_t)g * PNNS + j) * C + c]);
    out[t] = (h16)m;
}

// 3-NN + normalized inverse-distance weights (first-index tie-break).
__global__ void pn2_knn3(const float* __restrict__ xyz1, const float* __restrict__ xyz2,
                         int N1, int N2, int* __restrict__ idx3, float* __restrict__ w3, int B)
{
    int t = blockIdx.x * blockDim.x + threadIdx.x;
    if (t >= B * N1) return;
    int b = t / N1;
    const float px = xyz1[t * 3 + 0], py = xyz1[t * 3 + 1], pz = xyz1[t * 3 + 2];
    const float* q = xyz2 + (size_t)b * N2 * 3;
    float d0 = 1e30f, d1 = 1e30f, d2 = 1e30f;
    int   i0 = 0,     i1 = 0,     i2 = 0;
    for (int i = 0; i < N2; ++i) {
        float dx = q[i * 3 + 0] - px, dy = q[i * 3 + 1] - py, dz = q[i * 3 + 2] - pz;
        float d = dx * dx + dy * dy + dz * dz;
        if (d < d0)      { d2 = d1; i2 = i1; d1 = d0; i1 = i0; d0 = d; i0 = i; }
        else if (d < d1) { d2 = d1; i2 = i1; d1 = d;  i1 = i; }
        else if (d < d2) { d2 = d;  i2 = i; }
    }
    float w0 = 1.f / (d0 + 1e-8f), w1 = 1.f / (d1 + 1e-8f), w2 = 1.f / (d2 + 1e-8f);
    float ws = w0 + w1 + w2;
    idx3[t * 3 + 0] = i0; idx3[t * 3 + 1] = i1; idx3[t * 3 + 2] = i2;
    w3[t * 3 + 0] = w0 / ws; w3[t * 3 + 1] = w1 / ws; w3[t * 3 + 2] = w2 / ws;
}

// FP GEMM input: X[row, 0..CTp) = [ sum_j w_j*f2[idx_j] (C2) | skip (Cs) | 0 ]
__global__ void pn2_fp_build(const h16* __restrict__ f2, const h16* __restrict__ skip,
                             const int* __restrict__ idx3, const float* __restrict__ w3,
                             int N1, int N2, int C2, int Cs, int CTp,
                             h16* __restrict__ X, int B)
{
    long long e = (long long)blockIdx.x * blockDim.x + threadIdx.x;
    long long total = (long long)B * N1 * CTp;
    if (e >= total) return;
    int c = (int)(e % CTp);
    long long row = e / CTp;
    if (c < C2) {
        int b = (int)(row / N1);
        const h16*   fb = f2 + (size_t)b * N2 * C2;
        const int*   ib = idx3 + row * 3;
        const float* wb = w3 + row * 3;
        X[e] = (h16)(wb[0] * (float)fb[(size_t)ib[0] * C2 + c]
                   + wb[1] * (float)fb[(size_t)ib[1] * C2 + c]
                   + wb[2] * (float)fb[(size_t)ib[2] * C2 + c]);
    } else if (c < C2 + Cs) {
        X[e] = skip[row * Cs + (c - C2)];
    } else {
        X[e] = (h16)0.f;
    }
}

// skip0 = [one-hot class (16) | xyz (3) | stem feats (32)] (f16, stride 51)
__global__ void pn2_build_skip0(const float* __restrict__ cls, const float* __restrict__ xyz,
                                const h16* __restrict__ f0, h16* __restrict__ skip, int B, int N)
{
    const int Cs = PNOBJ + 3 + 32;
    long long e = (long long)blockIdx.x * blockDim.x + threadIdx.x;
    if (e >= (long long)B * N * Cs) return;
    int c = (int)(e % Cs);
    long long row = e / Cs;
    int b = (int)(row / N);
    if (c < PNOBJ)          skip[e] = (h16)cls[b * PNOBJ + c];
    else if (c < PNOBJ + 3) skip[e] = (h16)xyz[row * 3 + (c - PNOBJ)];
    else                    skip[e] = f0[row * 32 + (c - PNOBJ - 3)];
}

// BatchNorm batch stats (one block per channel) + apply/relu.
__global__ void __launch_bounds__(256)
pn2_bn_stats(const h16* __restrict__ H, int M, int C, float* __restrict__ mu, float* __restrict__ var)
{
    __shared__ float ssum[256], ssq[256];
    const int c = blockIdx.x, t = threadIdx.x;
    float s = 0.f, q = 0.f;
    for (int i = t; i < M; i += blockDim.x) {
        float v = (float)H[(size_t)i * C + c];
        s += v; q += v * v;
    }
    ssum[t] = s; ssq[t] = q;
    __syncthreads();
    for (int k = blockDim.x >> 1; k > 0; k >>= 1) {
        if (t < k) { ssum[t] += ssum[t + k]; ssq[t] += ssq[t + k]; }
        __syncthreads();
    }
    if (t == 0) {
        float m = ssum[0] / (float)M;
        mu[c]  = m;
        var[c] = ssq[0] / (float)M - m * m;
    }
}

__global__ void pn2_bn_apply_relu(h16* __restrict__ H, int M, int C,
                                  const float* __restrict__ mu, const float* __restrict__ var,
                                  const float* __restrict__ gamma, const float* __restrict__ beta)
{
    long long e = (long long)blockIdx.x * blockDim.x + threadIdx.x;
    if (e >= (long long)M * C) return;
    int c = (int)(e % C);
    float v = ((float)H[e] - mu[c]) * rsqrtf(var[c] + PN_BNEPS) * gamma[c] + beta[c];
    H[e] = (h16)fmaxf(v, 0.f);
}

// (B*N, C) f16 (row stride Cp) -> (B, C, N) f32
__global__ void pn2_to_bcn(const h16* __restrict__ in, float* __restrict__ out,
                           int B, int N, int C, int Cp)
{
    long long e = (long long)blockIdx.x * blockDim.x + threadIdx.x;
    if (e >= (long long)B * N * C) return;
    int c = (int)(e % C);
    long long row = e / C;
    int n = (int)(row % N);
    int b = (int)(row / N);
    out[((size_t)b * C + c) * N + n] = (float)in[row * Cp + c];
}

// ============================================================================
// Host orchestration
// ============================================================================
extern "C" void kernel_launch(void* const* d_in, const int* in_sizes, int n_in,
                              void* d_out, int out_size, void* d_ws, size_t ws_size,
                              hipStream_t stream)
{
    (void)in_sizes; (void)n_in; (void)out_size; (void)ws_size;
    const float* P[44];
    for (int i = 0; i < 44; ++i) P[i] = (const float*)d_in[i];

    const int B = PNB, N = PNN;
    const int NP[4]   = {1024, 256, 64, 16};
    const float RR[4] = {0.2f, 0.4f, 0.8f, 1.2f};
    const int SA_CI[4] = {32, 64, 128, 256};
    const int SA_CO[4] = {64, 128, 256, 512};

    // ---- workspace bump allocator (bytes, 256B aligned) -------------------
    char* base = (char*)d_ws;
    size_t off = 0;
    auto allocB = [&](size_t bytes) -> void* {
        void* p = base + off;
        off += (bytes + 255) & ~(size_t)255;
        return p;
    };
    auto allocF = [&](size_t n) { return (float*)allocB(n * 4); };
    auto allocH = [&](size_t n) { return (h16*)allocB(n * 2); };
    auto allocI = [&](size_t n) { return (int*)allocB(n * 4); };

    float* xyz0 = allocF((size_t)B * N * 3);
    h16*   xyzp = allocH((size_t)B * N * 32);
    h16*   f0   = allocH((size_t)B * N * 32);
    float* xl[4]; h16* fl[4];
    for (int i = 0; i < 4; ++i) {
        xl[i] = allocF((size_t)B * NP[i] * 3);
        fl[i] = allocH((size_t)B * NP[i] * SA_CO[i]);
    }
    h16* g3 = allocH((size_t)B * 64 * 256);
    h16* g2 = allocH((size_t)B * 256 * 128);
    h16* g1 = allocH((size_t)B * 1024 * 64);
    h16* g0 = allocH((size_t)B * 4096 * 32);
    h16* hfin   = allocH((size_t)B * N * 128);
    h16* logits = allocH((size_t)B * N * 64);      // Np(50) = 64
    float* mu  = allocF(128);
    float* var = allocF(128);
    int*   fidx = allocI((size_t)B * 1024);
    int*   gidx = allocI((size_t)B * 1024 * PNNS);
    int*   idx3 = allocI((size_t)B * N * 3);
    float* w3   = allocF((size_t)B * N * 3);
    h16* scrX  = allocH((size_t)B * 1024 * PNNS * 64);   // max Kp panel (SA1: 64)
    h16* scrH1 = allocH((size_t)B * 1024 * PNNS * 64);   // max mid (SA1: 64)
    h16* scrH2 = allocH((size_t)B * 1024 * PNNS * 64);   // max out / skip0

    // ---- pack all GEMM weights/biases into fragment layout ----------------
    struct WDesc { int wi, K, Nn; };
    const WDesc wd[20] = {
        {2, 3, 32},    {4, 32, 32},                      // stem
        {6, 35, 64},   {8, 64, 64},                      // sa1
        {10, 67, 128}, {12, 128, 128},                   // sa2
        {14, 131, 256},{16, 256, 256},                   // sa3
        {18, 259, 512},{20, 512, 512},                   // sa4
        {22, 768, 256},{24, 256, 256},                   // fp1
        {26, 384, 128},{28, 128, 128},                   // fp2
        {30, 192, 64}, {32, 64, 64},                     // fp3
        {34, 115, 32}, {36, 32, 32},                     // fp4
        {38, 32, 128}, {42, 128, 50},                    // fin
    };
    h16* wp[20]; float* bp[20]; int kp[20], np[20];
    for (int i = 0; i < 20; ++i) {
        kp[i] = pn_r32(wd[i].K);
        np[i] = pn_r32(wd[i].Nn);
        wp[i] = allocH((size_t)kp[i] * np[i]);
        bp[i] = allocF((size_t)np[i]);
        int tot = (kp[i] >> 5) * (np[i] >> 4) * 512;
        pn2_pack_w<<<dim3((tot + 255) / 256), 256, 0, stream>>>(
            P[wd[i].wi], wd[i].K, wd[i].Nn, kp[i], np[i], wp[i]);
        pn2_pack_b<<<dim3((np[i] + 63) / 64), 64, 0, stream>>>(P[wd[i].wi + 1], wd[i].Nn, np[i], bp[i]);
    }

    auto ewg = [](long long total) { return dim3((unsigned)((total + 255) / 256)); };
    auto gemm = [&](const h16* X, int widx, h16* Y, int M, int relu) {
        dim3 grid(np[widx] >> 5, M >> 6);                 // 16x32 strip / wave
        pn2_gemm<<<grid, dim3(128), 0, stream>>>(X, wp[widx], bp[widx], Y,
                                                 M, np[widx], kp[widx], relu);
    };

    // ---- stem -------------------------------------------------------------
    pn2_points_to_xyz<<<ewg((long long)B * N), 256, 0, stream>>>(P[0], xyz0, xyzp, B, N);
    gemm(xyzp,  0, scrH1, B * N, 1);
    gemm(scrH1, 1, f0,    B * N, 1);

    // ---- SA hierarchy -----------------------------------------------------
    const float* cur_xyz = xyz0; const h16* cur_f = f0; int curN = N;
    for (int L = 0; L < 4; ++L) {
        const int S = NP[L], Ci = SA_CI[L], Co = SA_CO[L];
        const int Kpp = pn_r32(Ci + 3);
        pn2_fps<<<B, 256, 0, stream>>>(cur_xyz, curN, S, fidx);
        pn2_gather_newxyz<<<ewg((long long)B * S * 3), 256, 0, stream>>>(cur_xyz, fidx, curN, S, xl[L], B);
        pn2_ball_query<<<ewg((long long)B * S), 256, 0, stream>>>(cur_xyz, xl[L], curN, S,
                                                                  RR[L] * RR[L], gidx, B);
        pn2_group_build<<<ewg((long long)B * S * PNNS * Kpp), 256, 0, stream>>>(
            cur_xyz, xl[L], cur_f, gidx, curN, S, Ci, Kpp, scrX, B);
        gemm(scrX,  2 + 2 * L, scrH1, B * S * PNNS, 1);
        gemm(scrH1, 3 + 2 * L, scrH2, B * S * PNNS, 1);
        pn2_maxpool<<<ewg((long long)B * S * Co), 256, 0, stream>>>(scrH2, B * S, Co, fl[L]);
        cur_xyz = xl[L]; cur_f = fl[L]; curN = S;
    }

    // ---- FP hierarchy -----------------------------------------------------
    struct FpCfg { const float *x1, *x2; const h16 *skip, *fc; int N1, Cs, N2, C2, wi; h16* out; };
    FpCfg fcfg[3] = {
        { xl[2], xl[3], fl[2], fl[3], 64,   256, 16,  512, 10, g3 },
        { xl[1], xl[2], fl[1], g3,    256,  128, 64,  256, 12, g2 },
        { xl[0], xl[1], fl[0], g2,    1024, 64,  256, 128, 14, g1 },
    };
    for (int L = 0; L < 3; ++L) {
        FpCfg c = fcfg[L];
        const int CTp = pn_r32(c.C2 + c.Cs);
        pn2_knn3<<<ewg((long long)B * c.N1), 256, 0, stream>>>(c.x1, c.x2, c.N1, c.N2, idx3, w3, B);
        pn2_fp_build<<<ewg((long long)B * c.N1 * CTp), 256, 0, stream>>>(
            c.fc, c.skip, idx3, w3, c.N1, c.N2, c.C2, c.Cs, CTp, scrX, B);
        gemm(scrX,  c.wi,     scrH1, B * c.N1, 1);
        gemm(scrH1, c.wi + 1, c.out, B * c.N1, 1);
    }
    // fp4: interpolate g1 (at x1) onto full cloud; skip0 = [cls|xyz|f0] (51 ch)
    const int Cs0 = PNOBJ + 3 + 32;
    const int CTp4 = pn_r32(64 + Cs0);                    // 115 -> 128
    pn2_build_skip0<<<ewg((long long)B * N * Cs0), 256, 0, stream>>>(P[1], xyz0, f0, scrH2, B, N);
    pn2_knn3<<<ewg((long long)B * N), 256, 0, stream>>>(xyz0, xl[0], N, 1024, idx3, w3, B);
    pn2_fp_build<<<ewg((long long)B * N * CTp4), 256, 0, stream>>>(
        g1, scrH2, idx3, w3, N, 1024, 64, Cs0, CTp4, scrX, B);
    gemm(scrX,  16, scrH1, B * N, 1);
    gemm(scrH1, 17, g0,    B * N, 1);

    // ---- final MLP + BN ---------------------------------------------------
    gemm(g0, 18, hfin, B * N, 0);
    pn2_bn_stats<<<128, 256, 0, stream>>>(hfin, B * N, 128, mu, var);
    pn2_bn_apply_relu<<<ewg((long long)B * N * 128), 256, 0, stream>>>(hfin, B * N, 128,
                                                                       mu, var, P[40], P[41]);
    gemm(hfin, 19, logits, B * N, 0);

    // ---- outputs: logits (B,50,N) ++ g0 (B,32,N), f32 ---------------------
    float* out = (float*)d_out;
    pn2_to_bcn<<<ewg((long long)B * N * PNCLS), 256, 0, stream>>>(logits, out, B, N, PNCLS, 64);
    pn2_to_bcn<<<ewg((long long)B * N * 32), 256, 0, stream>>>(g0, out + (size_t)B * PNCLS * N, B, N, 32, 32);
}
